// MLA_49340584296784
// MI455X (gfx1250) — matrix operand order repaired and checked by reading, not compiled, not run on hardware
//
#include <hip/hip_runtime.h>
#include <hip/hip_bf16.h>

#ifndef USE_ASYNC_COPY
#define USE_ASYNC_COPY 1   // gfx1250 async global->LDS copies (ASYNCcnt path)
#endif

typedef __bf16 bf16;
typedef __bf16 bf16x2  __attribute__((ext_vector_type(2)));
typedef __bf16 bf16x8  __attribute__((ext_vector_type(8)));
typedef __bf16 bf16x16 __attribute__((ext_vector_type(16)));
typedef float  f32x8   __attribute__((ext_vector_type(8)));

#define WMMA_BF16(a, b, c) \
  __builtin_amdgcn_wmma_f32_16x16x32_bf16(false, (a), false, (b), (short)0, (c), false, false)

__device__ __forceinline__ bf16x16 cat8(bf16x8 lo, bf16x8 hi) {
  return __builtin_shufflevector(lo, hi, 0,1,2,3,4,5,6,7,8,9,10,11,12,13,14,15);
}
__device__ __forceinline__ bf16x8 ld8(const bf16* p) { return *(const bf16x8*)p; }

// 16-byte async copy global -> LDS (per-lane), tracked by ASYNCcnt.
__device__ __forceinline__ void async_cp16(void* lds, const void* g) {
#if USE_ASYNC_COPY
  unsigned ldsOff = (unsigned)(unsigned long long)lds;   // low 32 bits = LDS offset
  asm volatile("global_load_async_to_lds_b128 %0, %1, off"
               :: "v"(ldsOff), "v"(g) : "memory");
#else
  *(bf16x8*)lds = *(const bf16x8*)g;
#endif
}
__device__ __forceinline__ void async_wait() {
#if USE_ASYNC_COPY
  asm volatile("s_wait_asynccnt 0x0" ::: "memory");
#endif
}

// ---------------------------------------------------------------------------
// Generic batched GEMM: C[m][n] = sum_k A[m][k] * B[n][k] (+bias[n])
// A,B bf16; C f32 or bf16. Tiles: BM=128, BN=64, BK=32. 256 thr = 8 wave32.
// Wave w: row tiles (w&3)*16 and (w&3)*16+64; col tiles (w>>2)*32 (+16).
// Double-buffered LDS, async global->LDS staging overlapped with WMMA.
// ---------------------------------------------------------------------------
__global__ __launch_bounds__(256)
void gemm_bf16_wmma(const bf16* __restrict__ A, const bf16* __restrict__ B,
                    float* __restrict__ Cf, bf16* __restrict__ Cb,
                    const float* __restrict__ bias,
                    int K, int lda, int ldb, int ldc,
                    long long aB, long long bB, long long cB) {
  __shared__ __align__(16) bf16 As[2][128 * 32];
  __shared__ __align__(16) bf16 Bs[2][64 * 32];

  const int tid  = threadIdx.x;
  const int lane = tid & 31;
  const int wave = tid >> 5;
  const int m0 = blockIdx.y * 128;
  const int n0 = blockIdx.x * 64;
  const long long z = blockIdx.z;
  A += z * aB;
  B += z * bB;

  const int mrow = (wave & 3) * 16;
  const int nc0  = (wave >> 2) * 32;
  const int half = lane >> 4;
  const int l16  = lane & 15;

  // staging assignment: thread -> one 16B chunk of B, two 16B chunks of A
  const int srow = tid >> 2;          // 0..63
  const int sk   = (tid & 3) * 8;     // 0,8,16,24

  f32x8 c00 = {}, c01 = {}, c10 = {}, c11 = {};

  // ---- prologue: stage tile 0 into buffer 0
  {
    const bf16* a0p = &A[(long long)(m0 + srow) * lda + sk];
    async_cp16(&As[0][srow * 32 + sk], a0p);
    async_cp16(&As[0][(srow + 64) * 32 + sk], a0p + (long long)64 * lda);
    async_cp16(&Bs[0][srow * 32 + sk], &B[(long long)(n0 + srow) * ldb + sk]);
  }
  async_wait();
  __syncthreads();

  int buf = 0;
  for (int k0 = 0; k0 < K; k0 += 32) {
    // ---- stage next tile into the other buffer (overlapped with WMMA below)
    const int kn = k0 + 32;
    if (kn < K) {
      const bf16* a0p = &A[(long long)(m0 + srow) * lda + kn + sk];
      async_cp16(&As[buf ^ 1][srow * 32 + sk], a0p);
      async_cp16(&As[buf ^ 1][(srow + 64) * 32 + sk], a0p + (long long)64 * lda);
      async_cp16(&Bs[buf ^ 1][srow * 32 + sk],
                 &B[(long long)(n0 + srow) * ldb + kn + sk]);
    }
    if (k0 + 64 < K) {  // warm GL2 for the tile after next
      __builtin_prefetch(&A[(long long)(m0 + srow) * lda + k0 + 64 + sk], 0, 3);
      __builtin_prefetch(&B[(long long)(n0 + srow) * ldb + k0 + 64 + sk], 0, 3);
    }

    // ---- compute from current buffer
    const bf16* as = As[buf];
    const bf16* bs = Bs[buf];
    bf16x16 a0 = cat8(ld8(&as[(mrow + l16) * 32 + half * 8]),
                      ld8(&as[(mrow + l16) * 32 + 16 + half * 8]));
    bf16x16 a1 = cat8(ld8(&as[(mrow + 64 + l16) * 32 + half * 8]),
                      ld8(&as[(mrow + 64 + l16) * 32 + 16 + half * 8]));
    bf16x16 b0 = cat8(ld8(&bs[(nc0 + l16) * 32 + half * 8]),
                      ld8(&bs[(nc0 + l16) * 32 + 16 + half * 8]));
    bf16x16 b1 = cat8(ld8(&bs[(nc0 + 16 + l16) * 32 + half * 8]),
                      ld8(&bs[(nc0 + 16 + l16) * 32 + 16 + half * 8]));
    c00 = WMMA_BF16(a0, b0, c00);
    c01 = WMMA_BF16(a0, b1, c01);
    c10 = WMMA_BF16(a1, b0, c10);
    c11 = WMMA_BF16(a1, b1, c11);

    async_wait();      // own async copies for next tile have landed
    __syncthreads();   // everyone done reading buf & writing buf^1
    buf ^= 1;
  }

  // ---- epilogue
  for (int rt = 0; rt < 2; ++rt) {
    for (int ct = 0; ct < 2; ++ct) {
      f32x8 c = rt ? (ct ? c11 : c10) : (ct ? c01 : c00);
      int n = n0 + nc0 + ct * 16 + l16;
      float bv = bias ? bias[n] : 0.0f;
      for (int r = 0; r < 8; ++r) {
        int m = m0 + mrow + rt * 64 + r + 8 * half;
        float v = c[r] + bv;
        long long idx = cB * z + (long long)m * ldc + n;
        if (Cb) Cb[idx] = (bf16)v; else Cf[idx] = v;
      }
    }
  }
}

// ---------------------------------------------------------------------------
// Reductions
// ---------------------------------------------------------------------------
__device__ __forceinline__ float waveRedSum(float v) {
  for (int off = 16; off > 0; off >>= 1) v += __shfl_xor(v, off, 32);
  return v;
}
__device__ __forceinline__ float waveRedMax(float v) {
  for (int off = 16; off > 0; off >>= 1) v = fmaxf(v, __shfl_xor(v, off, 32));
  return v;
}
__device__ __forceinline__ float blockRedSum(float v) {
  __shared__ float red[8];
  v = waveRedSum(v);
  if ((threadIdx.x & 31) == 0) red[threadIdx.x >> 5] = v;
  __syncthreads();
  float t = red[0] + red[1] + red[2] + red[3] + red[4] + red[5] + red[6] + red[7];
  __syncthreads();
  return t;
}

// ---------------------------------------------------------------------------
// Elementwise / epilogue kernels
// ---------------------------------------------------------------------------
__global__ __launch_bounds__(256)
void cvt_f32_bf16(const float* __restrict__ in, bf16* __restrict__ out, long long n) {
  long long i = (long long)blockIdx.x * 256 + threadIdx.x;
  long long stride = (long long)gridDim.x * 256;
  for (; i < n; i += stride) out[i] = (bf16)in[i];
}

// out[h][c][d] = wkb[(h*256+d)*512 + c] for d<128  (transpose of wkv_b[:, :128, :])
__global__ __launch_bounds__(256)
void cvt_wkb_transpose(const float* __restrict__ wkb, bf16* __restrict__ out) {
  long long i = (long long)blockIdx.x * 256 + threadIdx.x;
  if (i < (long long)16 * 512 * 128) {
    int d = (int)(i & 127);
    long long t = i >> 7;
    int c = (int)(t & 511);
    int h = (int)(t >> 9);
    out[i] = (bf16)wkb[((long long)(h * 256 + d)) * 512 + c];
  }
}

__global__ __launch_bounds__(256)
void rmsnorm_bf16(const float* __restrict__ in, const float* __restrict__ w,
                  bf16* __restrict__ out, int N) {
  long long row = blockIdx.x;
  const float* x = in + row * N;
  float ss = 0.0f;
  for (int i = threadIdx.x; i < N; i += 256) { float v = x[i]; ss += v * v; }
  float tot = blockRedSum(ss);
  float r = rsqrtf(tot / (float)N + 1e-6f);
  bf16* o = out + row * N;
  for (int i = threadIdx.x; i < N; i += 256) o[i] = (bf16)(x[i] * r * w[i]);
}

// kv_all [2048][576] -> kvn bf16 [2048][512] (rmsnorm), kpe bf16 [2048][64] (RoPE)
__global__ __launch_bounds__(256)
void kv_prep(const float* __restrict__ kv_all, const float* __restrict__ kvw,
             const float* __restrict__ cosT, const float* __restrict__ sinT,
             bf16* __restrict__ kvn, bf16* __restrict__ kpe) {
  long long row = blockIdx.x;
  int s = (int)(row & 1023);
  const float* x = kv_all + row * 576;
  float ss = 0.0f;
  for (int i = threadIdx.x; i < 512; i += 256) { float v = x[i]; ss += v * v; }
  float tot = blockRedSum(ss);
  float r = rsqrtf(tot / 512.0f + 1e-6f);
  for (int i = threadIdx.x; i < 512; i += 256) kvn[row * 512 + i] = (bf16)(x[i] * r * kvw[i]);
  if (threadIdx.x < 64) {
    int d = threadIdx.x;
    float v = x[512 + d];
    float other = x[512 + (d < 32 ? d + 32 : d - 32)];
    float rot = (d < 32) ? -other : other;
    kpe[row * 64 + d] = (bf16)(v * cosT[s * 64 + d] + rot * sinT[s * 64 + d]);
  }
}

// q f32 [2048][3072] -> bf16 with RoPE on each head's pe segment (d in [128,192))
__global__ __launch_bounds__(256)
void q_rope_cvt(const float* __restrict__ q, const float* __restrict__ cosT,
                const float* __restrict__ sinT, bf16* __restrict__ out) {
  long long row = blockIdx.x;
  int s = (int)(row & 1023);
  const float* x = q + row * 3072;
  bf16* o = out + row * 3072;
  for (int j = threadIdx.x; j < 3072; j += 256) {
    int d = j % 192;
    float v = x[j];
    if (d < 128) {
      o[j] = (bf16)v;
    } else {
      int dp = d - 128;
      float other = x[j - dp + (dp < 32 ? dp + 32 : dp - 32)];
      float rot = (dp < 32) ? -other : other;
      o[j] = (bf16)(v * cosT[s * 64 + dp] + rot * sinT[s * 64 + dp]);
    }
  }
}

// ---------------------------------------------------------------------------
// Fused flash-style MLA attention.
// Q tile = 32 queries, K tile = 32 keys per iteration, head dim 576 = 18 WMMA
// k-steps (16 from q_abs/kvn + 2 from q_pe/k_pe), V dim 512.
// 8 waves: scores -> (rowTile, keyTile, featureHalf); O -> (rowTile, 128-col blk).
// Online softmax distributed over all 8 waves (wave w owns rows 4w..4w+3,
// 32 lanes cover the 32 key columns; shfl trees for max/sum; row state in LDS).
// ---------------------------------------------------------------------------
__global__ __launch_bounds__(256)
void mla_attention(const bf16* __restrict__ qabs, const bf16* __restrict__ qbf,
                   const bf16* __restrict__ kvn, const bf16* __restrict__ kpe,
                   bf16* __restrict__ o512) {
  const int q0 = blockIdx.x * 32;
  const int b  = blockIdx.y;
  const int h  = blockIdx.z;
  const int tid  = threadIdx.x;
  const int lane = tid & 31;
  const int wave = tid >> 5;
  const int half = lane >> 4;
  const int l16  = lane & 15;

  __shared__ __align__(16) float ScoreP[2][32 * 32];
  __shared__ __align__(16) bf16  Ps[32 * 32];
  __shared__ __align__(16) bf16  Vt[512 * 32];
  __shared__ float alphaRow[32];
  __shared__ float rowMs[32];
  __shared__ float rowLs[32];

  const long long rowBase = (long long)b * 1024;

  const int s_rt = wave & 1;
  const int s_ct = (wave >> 1) & 1;
  const int s_fh = wave >> 2;
  const int o_rt = wave & 1;
  const int o_cb = wave >> 1;   // 0..3 -> cols cb*128

  f32x8 acc[8];
  for (int i = 0; i < 8; ++i) acc[i] = f32x8{};

  if (tid < 32) { rowMs[tid] = -3.0e38f; rowLs[tid] = 0.0f; }

  for (int t0 = 0; t0 <= q0; t0 += 32) {
    __syncthreads();
    // ---- stage V^T into LDS: Vt[c][t] = kvn[rowBase+t0+t][c]
    // two adjacent key rows per chunk -> packed bf16x2 stores (t, t+1 adjacent)
    for (int i = 0; i < 4; ++i) {
      int idx = tid + i * 256;            // 0..1023
      int tp  = idx >> 6;                 // 0..15 pair of key rows
      int cvec = (idx & 63) * 8;
      const bf16* src = &kvn[(rowBase + t0 + tp * 2) * 512 + cvec];
      bf16x8 r0 = ld8(src);
      bf16x8 r1 = ld8(src + 512);
      for (int e = 0; e < 8; ++e) {
        bf16x2 pr = { r0[e], r1[e] };
        *(bf16x2*)&Vt[(cvec + e) * 32 + tp * 2] = pr;
      }
    }
    // ---- partial scores for this wave's (rt, ct, feature-half)
    {
      f32x8 sc = {};
      long long qm = rowBase + q0 + s_rt * 16 + l16;
      long long km = rowBase + t0 + s_ct * 16 + l16;
      for (int st = s_fh * 9; st < s_fh * 9 + 9; ++st) {
        bf16x16 a, bb;
        if (st < 16) {
          const bf16* qa = &qabs[(qm * 16 + h) * 512 + st * 32];
          a  = cat8(ld8(&qa[half * 8]), ld8(&qa[16 + half * 8]));
          const bf16* kb = &kvn[km * 512 + st * 32 + half * 8];
          bb = cat8(ld8(&kb[0]), ld8(&kb[16]));
        } else {
          int f = (st - 16) * 32;
          const bf16* qa = &qbf[qm * 3072 + h * 192 + 128 + f];
          a  = cat8(ld8(&qa[half * 8]), ld8(&qa[16 + half * 8]));
          const bf16* kb = &kpe[km * 64 + f + half * 8];
          bb = cat8(ld8(&kb[0]), ld8(&kb[16]));
        }
        sc = WMMA_BF16(a, bb, sc);
      }
      for (int r = 0; r < 8; ++r) {
        int rr = s_rt * 16 + r + 8 * half;
        int cc = s_ct * 16 + l16;
        ScoreP[s_fh][rr * 32 + cc] = sc[r];
      }
    }
    __syncthreads();
    // ---- online softmax: wave w owns rows 4w..4w+3; lanes cover 32 key cols
    for (int rr = 0; rr < 4; ++rr) {
      int r = wave * 4 + rr;
      int q = q0 + r;
      float s = (ScoreP[0][r * 32 + lane] + ScoreP[1][r * 32 + lane])
                * 0.07216878364870323f;
      if (t0 + lane > q) s += -1.0e9f;
      float mOld = rowMs[r];
      float mNew = fmaxf(mOld, waveRedMax(s));
      float p = __expf(s - mNew);
      float psum = waveRedSum(p);
      float alpha = __expf(mOld - mNew);
      Ps[r * 32 + lane] = (bf16)p;
      if (lane == 0) {
        rowMs[r] = mNew;
        rowLs[r] = rowLs[r] * alpha + psum;
        alphaRow[r] = alpha;
      }
    }
    __syncthreads();
    // ---- O update: rescale accumulators, then P @ V (one 32-deep WMMA step)
    {
      int rbase = o_rt * 16;
      int arow = rbase + l16;
      bf16x16 a = cat8(ld8(&Ps[arow * 32 + half * 8]),
                       ld8(&Ps[arow * 32 + 16 + half * 8]));
      for (int ct = 0; ct < 8; ++ct) {
        f32x8 c = acc[ct];
        for (int e = 0; e < 8; ++e) c[e] *= alphaRow[rbase + e + 8 * half];
        int col = o_cb * 128 + ct * 16 + l16;
        bf16x16 bb = cat8(ld8(&Vt[col * 32 + half * 8]),
                          ld8(&Vt[col * 32 + 16 + half * 8]));
        acc[ct] = WMMA_BF16(a, bb, c);
      }
    }
  }
  // ---- finalize: divide by l and write o512[m][h][c]
  __syncthreads();
  for (int ct = 0; ct < 8; ++ct) {
    int col = o_cb * 128 + ct * 16 + l16;
    for (int e = 0; e < 8; ++e) {
      int rr = o_rt * 16 + e + 8 * half;
      long long m = rowBase + q0 + rr;
      o512[(m * 16 + h) * 512 + col] = (bf16)(acc[ct][e] / rowLs[rr]);
    }
  }
}

// ---------------------------------------------------------------------------
// Host orchestration
// ---------------------------------------------------------------------------
static void launch_gemm(const bf16* A, const bf16* B, float* Cf, bf16* Cb,
                        const float* bias, int M, int N, int K,
                        int lda, int ldb, int ldc,
                        long long aB, long long bB, long long cB, int batches,
                        hipStream_t s) {
  dim3 grid(N / 64, M / 128, batches);
  gemm_bf16_wmma<<<grid, 256, 0, s>>>(A, B, Cf, Cb, bias, K, lda, ldb, ldc, aB, bB, cB);
}

extern "C" void kernel_launch(void* const* d_in, const int* in_sizes, int n_in,
                              void* d_out, int out_size, void* d_ws, size_t ws_size,
                              hipStream_t stream) {
  (void)in_sizes; (void)n_in; (void)out_size; (void)ws_size;

  const float* x        = (const float*)d_in[0];
  // d_in[1] = mask (pure causal -1e9; computed inline in the attention kernel)
  const float* wq_a_w   = (const float*)d_in[2];
  const float* wq_a_b   = (const float*)d_in[3];
  const float* q_norm_w = (const float*)d_in[4];
  const float* wq_b_w   = (const float*)d_in[5];
  const float* wq_b_b   = (const float*)d_in[6];
  const float* wkv_a_w  = (const float*)d_in[7];
  const float* wkv_a_b  = (const float*)d_in[8];
  const float* kv_norm_w= (const float*)d_in[9];
  const float* wk_b_w   = (const float*)d_in[10];
  const float* wo_w     = (const float*)d_in[11];
  const float* wo_b     = (const float*)d_in[12];
  const float* cosT     = (const float*)d_in[13];
  const float* sinT     = (const float*)d_in[14];

  // dims: B=2 S=1024 DIM=2048 H=16 QR=1536 KVR=512 DN=128 DR=64 DV=128
  const int M = 2048;                    // B*S rows

  char* W = (char*)d_ws;
  size_t off = 0;
  auto take = [&](size_t bytes) -> char* {
    char* p = W + off;
    off += (bytes + 255) & ~(size_t)255;
    return p;
  };
  bf16*  xbf     = (bf16*)take((size_t)M * 2048 * 2);
  bf16*  wqa_bf  = (bf16*)take((size_t)1536 * 2048 * 2);
  bf16*  wqb_bf  = (bf16*)take((size_t)3072 * 1536 * 2);
  bf16*  wkva_bf = (bf16*)take((size_t)576 * 2048 * 2);
  bf16*  wkb_bf  = (bf16*)take((size_t)4096 * 512 * 2);
  bf16*  wkbT_bf = (bf16*)take((size_t)16 * 512 * 128 * 2);
  bf16*  wo_bf   = (bf16*)take((size_t)2048 * 2048 * 2);
  float* qlat_f  = (float*)take((size_t)M * 1536 * 4);
  bf16*  qlat_bf = (bf16*)take((size_t)M * 1536 * 2);
  float* q_f     = (float*)take((size_t)M * 3072 * 4);
  bf16*  q_bf    = (bf16*)take((size_t)M * 3072 * 2);
  float* kva_f   = (float*)take((size_t)M * 576 * 4);
  bf16*  kvn_bf  = (bf16*)take((size_t)M * 512 * 2);
  bf16*  kpe_bf  = (bf16*)take((size_t)M * 64 * 2);
  bf16*  qabs_bf = (bf16*)take((size_t)M * 16 * 512 * 2);
  bf16*  o512_bf = (bf16*)take((size_t)M * 16 * 512 * 2);
  bf16*  ohead_bf= (bf16*)take((size_t)M * 2048 * 2);

  // 1) bf16 conversions
  cvt_f32_bf16<<<4096, 256, 0, stream>>>(x,       xbf,     (long long)M * 2048);
  cvt_f32_bf16<<<4096, 256, 0, stream>>>(wq_a_w,  wqa_bf,  (long long)1536 * 2048);
  cvt_f32_bf16<<<4096, 256, 0, stream>>>(wq_b_w,  wqb_bf,  (long long)3072 * 1536);
  cvt_f32_bf16<<<4096, 256, 0, stream>>>(wkv_a_w, wkva_bf, (long long)576 * 2048);
  cvt_f32_bf16<<<4096, 256, 0, stream>>>(wk_b_w,  wkb_bf,  (long long)4096 * 512);
  cvt_f32_bf16<<<4096, 256, 0, stream>>>(wo_w,    wo_bf,   (long long)2048 * 2048);
  cvt_wkb_transpose<<<4096, 256, 0, stream>>>(wk_b_w, wkbT_bf);

  // 2) q_lat = x @ wq_a^T + b  -> rmsnorm -> bf16
  launch_gemm(xbf, wqa_bf, qlat_f, nullptr, wq_a_b, M, 1536, 2048,
              2048, 2048, 1536, 0, 0, 0, 1, stream);
  rmsnorm_bf16<<<M, 256, 0, stream>>>(qlat_f, q_norm_w, qlat_bf, 1536);

  // 3) q = q_lat @ wq_b^T + b  -> RoPE(pe) + bf16
  launch_gemm(qlat_bf, wqb_bf, q_f, nullptr, wq_b_b, M, 3072, 1536,
              1536, 1536, 3072, 0, 0, 0, 1, stream);
  q_rope_cvt<<<M, 256, 0, stream>>>(q_f, cosT, sinT, q_bf);

  // 4) kv_all = x @ wkv_a^T + b -> rmsnorm(kv)->kvn, RoPE(k_pe)->kpe
  launch_gemm(xbf, wkva_bf, kva_f, nullptr, wkv_a_b, M, 576, 2048,
              2048, 2048, 576, 0, 0, 0, 1, stream);
  kv_prep<<<M, 256, 0, stream>>>(kva_f, kv_norm_w, cosT, sinT, kvn_bf, kpe_bf);

  // 5) q_abs[m,h,c] = sum_d q_nope[m,h,d] * wkv_b[h,d,c]   (B = W^T, batched over h)
  launch_gemm(q_bf, wkbT_bf, nullptr, qabs_bf, nullptr, M, 512, 128,
              3072, 128, 8192, /*aB=*/192, /*bB=*/(long long)512 * 128, /*cB=*/512,
              16, stream);

  // 6) fused attention (scores + online softmax + P@V), per (query-tile, b, h)
  {
    dim3 grid(32, 2, 16);
    mla_attention<<<grid, 256, 0, stream>>>(qabs_bf, q_bf, kvn_bf, kpe_bf, o512_bf);
  }

  // 7) per-head value projection: o[m,h,d] = sum_c o512[m,h,c] * wkv_b[h,128+d,c]
  launch_gemm(o512_bf, wkb_bf + (size_t)128 * 512, nullptr, ohead_bf, nullptr,
              M, 128, 512, 8192, 512, 2048,
              /*aB=*/512, /*bB=*/(long long)256 * 512, /*cB=*/128, 16, stream);

  // 8) out = o_heads @ wo^T + wo_b  (f32 to d_out)
  launch_gemm(ohead_bf, wo_bf, (float*)d_out, nullptr, wo_b, M, 2048, 2048,
              2048, 2048, 2048, 0, 0, 0, 1, stream);
}
